// Encoder_953482739902
// MI455X (gfx1250) — matrix operand order repaired and checked by reading, not compiled, not run on hardware
//
#include <hip/hip_runtime.h>
#include <math.h>

typedef __attribute__((ext_vector_type(16))) _Float16 v16h;
typedef __attribute__((ext_vector_type(8)))  _Float16 v8h;
typedef __attribute__((ext_vector_type(8)))  float    v8f;

#define CH 128
#define LDS_STRIDE 136          // 272 B row stride: 16B-aligned, conflict-free b128 lane spread
#define MT_PER_BLOCK 8

// ---------------- init: zero out, deg = 1 (self loop) ----------------
__global__ void k_init(float* __restrict__ out, float* __restrict__ deg,
                       int nOut, int nNodes) {
    int i = blockIdx.x * blockDim.x + threadIdx.x;
    int stride = gridDim.x * blockDim.x;
    for (int j = i; j < nOut; j += stride) out[j] = 0.0f;
    for (int j = i; j < nNodes; j += stride) deg[j] = 1.0f;
}

// ---------------- degree accumulation over edge sources (ei[0..E)) ----------------
__global__ void k_degree(const int* __restrict__ col, float* __restrict__ deg, int E) {
    int i = blockIdx.x * blockDim.x + threadIdx.x;
    int stride = gridDim.x * blockDim.x;
    for (int e = i; e < E; e += stride) {
        atomicAdd(&deg[col[e]], 1.0f);
    }
}

// ---------------- deg -> rsqrt(deg) in place (deg >= 1 always) ----------------
__global__ void k_dinv(float* __restrict__ deg, int n) {
    int i = blockIdx.x * blockDim.x + threadIdx.x;
    if (i < n) deg[i] = rsqrtf(deg[i]);
}

// ---------------- h = x @ W via v_wmma_f32_16x16x32_f16 ----------------
// One block = 8 waves. Block stages W^T (f16) in padded LDS once, loops 8 M-tiles.
// Wave w computes the 16x16 tile at (mtile, ntile = w), K = 128 in 4 WMMA steps.
__global__ __launch_bounds__(256) void k_gemm(const float* __restrict__ x,
                                              const float* __restrict__ W,
                                              float* __restrict__ h,
                                              int nNodes, int mtTotal) {
    __shared__ _Float16 lds_wT[CH * LDS_STRIDE];  // [n][k], padded rows
    __shared__ _Float16 lds_x[16 * LDS_STRIDE];   // [m][k], padded rows

    const int tid = threadIdx.x;

    // cooperative load of W, transposed to [n][k], converted to f16
    for (int idx = tid; idx < CH * CH; idx += 256) {
        int k = idx >> 7;
        int n = idx & (CH - 1);
        lds_wT[n * LDS_STRIDE + k] = (_Float16)W[idx];
    }
    __syncthreads();

    const int wave   = tid >> 5;
    const int lane   = tid & 31;
    const int l16    = lane & 15;        // row of A / col of B / col of D
    const int hiHalf = lane >> 4;        // lane group 0/1
    const int n0     = wave * 16;

    // B fragments: 16 contiguous K values per lane; fixed per wave -> hoist
    v16h bfrag[4];
#pragma unroll
    for (int kc = 0; kc < 4; ++kc) {
        const int kbB = kc * 32 + hiHalf * 16;
        const _Float16* wp = &lds_wT[(n0 + l16) * LDS_STRIDE + kbB];
        v8h blo = *(const v8h*)(wp);
        v8h bhi = *(const v8h*)(wp + 8);
        bfrag[kc] = __builtin_shufflevector(blo, bhi,
                    0,1,2,3,4,5,6,7,8,9,10,11,12,13,14,15);
    }

    for (int it = 0; it < MT_PER_BLOCK; ++it) {
        const int mt = blockIdx.x * MT_PER_BLOCK + it;
        if (mt >= mtTotal) break;

        __syncthreads();   // previous tile fully consumed
        for (int idx = tid; idx < 16 * CH; idx += 256) {
            int r = idx >> 7;
            int c = idx & (CH - 1);
            int row = mt * 16 + r;
            lds_x[r * LDS_STRIDE + c] =
                (row < nNodes) ? (_Float16)x[(size_t)row * CH + c] : (_Float16)0.0f;
        }
        __syncthreads();

        // Load all four A fragments first (overlap ds latency), then chain WMMAs.
        // A 16x32 f16 layout: lanes 0-15 hold K = kc*32 + {0..15},
        //                     lanes 16-31 hold K = kc*32 + {16..31}
        v16h afrag[4];
#pragma unroll
        for (int kc = 0; kc < 4; ++kc) {
            const int kbA = kc * 32 + hiHalf * 16;
            const _Float16* ap = &lds_x[l16 * LDS_STRIDE + kbA];
            v8h alo = *(const v8h*)(ap);
            v8h ahi = *(const v8h*)(ap + 8);
            afrag[kc] = __builtin_shufflevector(alo, ahi,
                        0,1,2,3,4,5,6,7,8,9,10,11,12,13,14,15);
        }

        v8f c = {};
#pragma unroll
        for (int kc = 0; kc < 4; ++kc) {
            c = __builtin_amdgcn_wmma_f32_16x16x32_f16(
                    false, afrag[kc], false, bfrag[kc], (short)0, c, false, false);
        }

        // D layout: VGPR r -> row (mt*16 + r + hiHalf*8), col = n0 + l16
        float* hp = h + ((size_t)(mt * 16 + hiHalf * 8)) * CH + n0 + l16;
        if (mt * 16 + 16 <= nNodes) {
            // full tile: unguarded stores at immediate offsets r*CH
#pragma unroll
            for (int r = 0; r < 8; ++r) hp[(size_t)r * CH] = c[r];
        } else {
#pragma unroll
            for (int r = 0; r < 8; ++r) {
                if (mt * 16 + hiHalf * 8 + r < nNodes) hp[(size_t)r * CH] = c[r];
            }
        }
    }
}

// ---------------- edge scatter: out[col] += h[row] * dinv[row]*dinv[col] ----------------
// Each wave loads 32 edges coalesced (one per lane), then broadcasts (row,col,norm)
// via shuffles while all 32 lanes move the 128-channel payload (float4 each).
__global__ __launch_bounds__(256) void k_scatter(const int* __restrict__ ei,
                                                 const float* __restrict__ h,
                                                 const float* __restrict__ dinv,
                                                 float* __restrict__ out, int E) {
    const int gwave  = (blockIdx.x * blockDim.x + threadIdx.x) >> 5;
    const int lane   = threadIdx.x & 31;
    const int nwaves = (gridDim.x * blockDim.x) >> 5;

    for (int ebase = gwave * 32; ebase < E; ebase += nwaves * 32) {
        const int myE = ebase + lane;
        int   rowL = 0, colL = 0;
        float normL = 0.0f;
        if (myE < E) {
            rowL  = ei[myE];
            colL  = ei[E + myE];
            normL = dinv[rowL] * dinv[colL];
        }
        const int cnt = min(32, E - ebase);
        for (int j = 0; j < cnt; ++j) {
            const int   row  = __shfl(rowL, j);
            const int   col  = __shfl(colL, j);
            const float norm = __shfl(normL, j);
            const float4 hv = *(const float4*)(h + (size_t)row * CH + lane * 4);
            float* op = out + (size_t)col * CH + lane * 4;
            atomicAdd(op + 0, hv.x * norm);
            atomicAdd(op + 1, hv.y * norm);
            atomicAdd(op + 2, hv.z * norm);
            atomicAdd(op + 3, hv.w * norm);
        }
    }
}

// ---------------- epilogue: self-loop + bias + mish ----------------
__device__ __forceinline__ float mishf(float v) {
    float sp = (v > 20.0f) ? v : log1pf(__expf(v));
    return v * tanhf(sp);
}

__global__ __launch_bounds__(256) void k_finish(float* __restrict__ out,
                                                const float* __restrict__ h,
                                                const float* __restrict__ dinv,
                                                const float* __restrict__ b,
                                                int nNodes) {
    const int total = nNodes * (CH / 4);
    const int start = blockIdx.x * blockDim.x + threadIdx.x;
    const int stride = gridDim.x * blockDim.x;
    for (int idx = start; idx < total; idx += stride) {
        const size_t base = (size_t)idx * 4;
        const int node = (int)(base >> 7);          // /128
        const int c0   = (int)(base & (CH - 1));
        const float di = dinv[node];
        const float s  = di * di;
        float4 hv = *(const float4*)(h + base);
        float4 ov = *(const float4*)(out + base);
        const float4 bv = *(const float4*)(b + c0);
        ov.x = mishf(ov.x + hv.x * s + bv.x);
        ov.y = mishf(ov.y + hv.y * s + bv.y);
        ov.z = mishf(ov.z + hv.z * s + bv.z);
        ov.w = mishf(ov.w + hv.w * s + bv.w);
        *(float4*)(out + base) = ov;
    }
}

extern "C" void kernel_launch(void* const* d_in, const int* in_sizes, int n_in,
                              void* d_out, int out_size, void* d_ws, size_t ws_size,
                              hipStream_t stream) {
    const float* x  = (const float*)d_in[0];
    const int*   ei = (const int*)d_in[1];
    const float* W  = (const float*)d_in[2];
    const float* b  = (const float*)d_in[3];
    float* out = (float*)d_out;

    const int nNodes  = in_sizes[0] / CH;
    const int E       = in_sizes[1] / 2;
    const int mtTotal = (nNodes + 15) / 16;

    float* ws  = (float*)d_ws;
    float* h   = ws;                               // nNodes * CH floats
    float* deg = ws + (size_t)nNodes * CH;         // nNodes floats (becomes dinv)

    k_init<<<1024, 256, 0, stream>>>(out, deg, nNodes * CH, nNodes);
    k_degree<<<2048, 256, 0, stream>>>(ei, deg, E);
    k_dinv<<<(nNodes + 255) / 256, 256, 0, stream>>>(deg, nNodes);
    k_gemm<<<(mtTotal + MT_PER_BLOCK - 1) / MT_PER_BLOCK, 256, 0, stream>>>(x, W, h, nNodes, mtTotal);
    k_scatter<<<4096, 256, 0, stream>>>(ei, h, deg, out, E);
    k_finish<<<4096, 256, 0, stream>>>(out, h, deg, b, nNodes);
}
